// RelativeMultiHeadAttention_70703751627040
// MI455X (gfx1250) — matrix operand order repaired and checked by reading, not compiled, not run on hardware
//
#include <hip/hip_runtime.h>
#include <hip/hip_bf16.h>

// ---------------- problem constants ----------------
#define D_MODEL 512
#define N_HEADS 16
#define D_HEAD  32
#define SEQ     2048
#define BATCH   2
#define BLK     64
#define NB      32          // SEQ / BLK
#define KKEEP   16          // ceil(0.5 * NB)
#define MROWS   (BATCH*SEQ) // 4096 GEMM rows

typedef __attribute__((ext_vector_type(16))) __bf16 v16bf;
typedef __attribute__((ext_vector_type(8)))  __bf16 v8bf;
typedef __attribute__((ext_vector_type(4)))  __bf16 v4bf;
typedef __attribute__((ext_vector_type(8)))  float  v8f;

__device__ __forceinline__ __bf16 f2bf(float f) { return (__bf16)f; }  // native cvt
__device__ __forceinline__ float  bf2f(__bf16 b) { return (float)b; }

// ================================================================
// Kernel 0: bulk f32 -> bf16 conversion (float4 in, 4x bf16 out).
// Hoists ALL conversions out of the GEMM inner loops.
// ================================================================
__global__ void k_cvt(const float* __restrict__ src, __bf16* __restrict__ dst, int n4)
{
  const int i = blockIdx.x * blockDim.x + threadIdx.x;
  if (i >= n4) return;
  const float4 v = ((const float4*)src)[i];
  v4bf o;
  o[0] = f2bf(v.x); o[1] = f2bf(v.y); o[2] = f2bf(v.z); o[3] = f2bf(v.w);
  ((v4bf*)dst)[i] = o;
}

// ================================================================
// Kernel A: C[M=4096, N=512] = X @ W^T + bias, stored head-split bf16.
// X, W pre-converted bf16. vtrans==0 -> dst[b][h][s][d]; 1 -> dst[b][h][d][s].
// WG: 128 threads (4 waves); wave w owns rows m0..m0+15, 4 N-tiles.
// ================================================================
__global__ __launch_bounds__(128)
void k_proj(const __bf16* __restrict__ X, const __bf16* __restrict__ W,
            const float* __restrict__ bias, __bf16* __restrict__ dst, int vtrans)
{
  const int wave = threadIdx.x >> 5;
  const int lane = threadIdx.x & 31;
  const int half = lane >> 4;
  const int ln   = lane & 15;
  const int m0 = blockIdx.x * 64 + wave * 16;
  const int n0 = blockIdx.y * 64;

  v8f acc[4];
  #pragma unroll
  for (int t = 0; t < 4; ++t)
    #pragma unroll
    for (int i = 0; i < 8; ++i) acc[t][i] = 0.0f;

  const __bf16* arow = X + (size_t)(m0 + ln) * D_MODEL;
  for (int kb = 0; kb < D_MODEL; kb += 32) {
    v8bf a0 = *(const v8bf*)(arow + kb + half * 16);
    v8bf a1 = *(const v8bf*)(arow + kb + half * 16 + 8);
    v16bf afrag = __builtin_shufflevector(a0, a1, 0,1,2,3,4,5,6,7,8,9,10,11,12,13,14,15);
    #pragma unroll
    for (int t = 0; t < 4; ++t) {
      const int n = n0 + t * 16 + ln;            // W row (B[k][n] = W[n][k])
      v16bf bfrag = *(const v16bf*)(W + (size_t)n * D_MODEL + kb + half * 16);
      acc[t] = __builtin_amdgcn_wmma_f32_16x16x32_bf16(
          false, afrag, false, bfrag, (short)0, acc[t], false, false);
    }
  }

  #pragma unroll
  for (int t = 0; t < 4; ++t) {
    const int n = n0 + t * 16 + ln;
    const float bv = bias[n];
    const int h = n >> 5, d = n & 31;
    #pragma unroll
    for (int j = 0; j < 8; ++j) {
      const int m = m0 + j + half * 8;           // global row = b*SEQ + s
      const int b = m >> 11, s = m & (SEQ - 1);
      const __bf16 o = f2bf(acc[t][j] + bv);
      if (!vtrans) dst[(((size_t)b * N_HEADS + h) * SEQ + s) * D_HEAD + d] = o;
      else         dst[(((size_t)b * N_HEADS + h) * D_HEAD + d) * SEQ + s] = o;
    }
  }
}

// ================================================================
// Kernel B: block means of Q and K over 64-token blocks -> f32
// ================================================================
__global__ void k_means(const __bf16* __restrict__ Qh, const __bf16* __restrict__ Kh,
                        float* __restrict__ qb, float* __restrict__ kb)
{
  const int idx = blockIdx.x * blockDim.x + threadIdx.x;   // [B*H*NB*Dh]
  if (idx >= BATCH * N_HEADS * NB * D_HEAD) return;
  const int d   = idx & 31;
  const int blk = (idx >> 5) & (NB - 1);
  const int bh  = idx >> 10;
  const __bf16* qp = Qh + ((size_t)bh * SEQ + blk * BLK) * D_HEAD + d;
  const __bf16* kp = Kh + ((size_t)bh * SEQ + blk * BLK) * D_HEAD + d;
  float sq = 0.f, sk = 0.f;
  #pragma unroll 8
  for (int t = 0; t < BLK; ++t) { sq += bf2f(qp[t * D_HEAD]); sk += bf2f(kp[t * D_HEAD]); }
  qb[idx] = sq * (1.0f / BLK);
  kb[idx] = sk * (1.0f / BLK);
}

// ================================================================
// Kernel C: block scores + causal top-KKEEP selection -> 32-bit mask/row
// One wave per (b,h); lane i owns block-row i.
// ================================================================
__global__ __launch_bounds__(32)
void k_mask(const float* __restrict__ qb, const float* __restrict__ kb,
            unsigned* __restrict__ bmask)
{
  __shared__ float sc[NB][NB];
  const int bh = blockIdx.x;
  const int i  = threadIdx.x;
  const float scale = 0.17677669529663687f;   // 1/sqrt(Dh)

  float qr[D_HEAD];
  #pragma unroll
  for (int d = 0; d < D_HEAD; ++d) qr[d] = qb[(bh * NB + i) * D_HEAD + d];

  for (int j = 0; j < NB; ++j) {
    const float* kr = kb + (bh * NB + j) * D_HEAD;
    float s = 0.f;
    #pragma unroll
    for (int d = 0; d < D_HEAD; ++d) s += qr[d] * kr[d];
    sc[i][j] = (j <= i) ? s * scale : -3.0e38f;
  }
  __syncthreads();

  unsigned mask = 0;
  const int ncand = i + 1;
  if (ncand <= KKEEP) {
    mask = (i == 31) ? 0xFFFFFFFFu : ((1u << ncand) - 1u);
  } else {
    for (int t = 0; t < KKEEP; ++t) {           // repeated max-extraction
      float best = -3.4e38f; int bj = 0;
      for (int j = 0; j <= i; ++j) { float v = sc[i][j]; if (v > best) { best = v; bj = j; } }
      mask |= (1u << bj);
      sc[i][bj] = -3.4e38f;
    }
  }
  mask |= (1u << i);                            // diagonal always kept
  const unsigned causal = (i == 31) ? 0xFFFFFFFFu : ((1u << (i + 1)) - 1u);
  bmask[bh * NB + i] = mask & causal;
}

// ================================================================
// Kernel D: block-sparse flash attention. WG = (b,h,qblock), 128 threads.
// Wave w owns query rows 16w..16w+15. Per kept KV block: 4 S-WMMAs + 4 PV-WMMAs.
// ================================================================
__global__ __launch_bounds__(128)
void k_attn(const __bf16* __restrict__ Qh, const __bf16* __restrict__ Kh,
            const __bf16* __restrict__ Vt, const unsigned* __restrict__ bmask,
            __bf16* __restrict__ ctx)
{
  __shared__ __attribute__((aligned(16))) float  Sbuf[64][68];
  __shared__ __attribute__((aligned(16))) __bf16 Pbuf[64][72];   // 144B rows, 16B aligned
  __shared__ float mrow[64], lrow[64], crow[64];

  const int wg   = blockIdx.x;
  const int qblk = wg & (NB - 1);
  const int bh   = wg >> 5;
  const int wave = threadIdx.x >> 5;
  const int lane = threadIdx.x & 31;
  const int half = lane >> 4;
  const int ln   = lane & 15;
  const float scale = 0.17677669529663687f;

  if (threadIdx.x < 64) { mrow[threadIdx.x] = -3.0e38f; lrow[threadIdx.x] = 0.0f; }
  __syncthreads();

  // Q A-fragment for this wave (rows qblk*64 + wave*16 + [0..15])
  const __bf16* qrow = Qh + ((size_t)bh * SEQ + qblk * 64 + wave * 16 + ln) * D_HEAD;
  v8bf q0 = *(const v8bf*)(qrow + half * 16);
  v8bf q1 = *(const v8bf*)(qrow + half * 16 + 8);
  v16bf aQ = __builtin_shufflevector(q0, q1, 0,1,2,3,4,5,6,7,8,9,10,11,12,13,14,15);

  v8f o0, o1;
  #pragma unroll
  for (int i = 0; i < 8; ++i) { o0[i] = 0.f; o1[i] = 0.f; }

  const unsigned mask = bmask[bh * NB + qblk];   // uniform -> scalar branch

  for (int j = 0; j < NB; ++j) {
    if (!((mask >> j) & 1u)) continue;

    // prefetch next KV block (gfx1250 global_prefetch path)
    if (j + 1 < NB) {
      __builtin_prefetch(Kh + ((size_t)bh * SEQ + (j + 1) * 64 + lane) * D_HEAD, 0, 1);
      __builtin_prefetch(Vt + ((size_t)bh * D_HEAD + lane) * SEQ + (j + 1) * 64, 0, 1);
    }

    // ---- S = Q K^T (4 N-subtiles of 16 KV tokens) ----
    #pragma unroll
    for (int t = 0; t < 4; ++t) {
      const int kvtok = j * 64 + t * 16 + ln;
      v16bf bK = *(const v16bf*)(Kh + ((size_t)bh * SEQ + kvtok) * D_HEAD + half * 16);
      v8f sacc;
      #pragma unroll
      for (int i = 0; i < 8; ++i) sacc[i] = 0.f;
      sacc = __builtin_amdgcn_wmma_f32_16x16x32_bf16(
          false, aQ, false, bK, (short)0, sacc, false, false);
      #pragma unroll
      for (int jj = 0; jj < 8; ++jj) {
        const int r = wave * 16 + jj + half * 8;
        const int c = t * 16 + ln;
        float v = sacc[jj] * scale;
        if (j == qblk && c > r) v = -3.0e38f;    // in-diagonal causal mask
        Sbuf[r][c] = v;
      }
    }
    __syncthreads();

    // ---- online softmax, one thread per query row ----
    if (threadIdx.x < 64) {
      const int r = threadIdx.x;
      const float mo = mrow[r];
      float mx = mo;
      #pragma unroll 8
      for (int c = 0; c < 64; ++c) mx = fmaxf(mx, Sbuf[r][c]);
      const float corr = __expf(mo - mx);
      float sum = 0.f;
      #pragma unroll 8
      for (int c = 0; c < 64; ++c) {
        const float p = __expf(Sbuf[r][c] - mx);
        sum += p;
        Pbuf[r][c] = f2bf(p);
      }
      mrow[r] = mx;
      lrow[r] = lrow[r] * corr + sum;
      crow[r] = corr;
    }
    __syncthreads();

    // ---- rescale O accumulators by per-row correction ----
    #pragma unroll
    for (int jj = 0; jj < 8; ++jj) {
      const float corr = crow[wave * 16 + jj + half * 8];
      o0[jj] *= corr; o1[jj] *= corr;
    }

    // ---- O += P @ V  (2 K-steps of 32, 2 N-tiles of 16) ----
    const __bf16* prow = &Pbuf[wave * 16 + ln][0];
    #pragma unroll
    for (int s = 0; s < 2; ++s) {
      v8bf p0 = *(const v8bf*)(prow + s * 32 + half * 16);
      v8bf p1 = *(const v8bf*)(prow + s * 32 + half * 16 + 8);
      v16bf aP = __builtin_shufflevector(p0, p1, 0,1,2,3,4,5,6,7,8,9,10,11,12,13,14,15);

      const size_t tokbase = (size_t)j * 64 + s * 32 + half * 16;
      v16bf bV0 = *(const v16bf*)(Vt + ((size_t)bh * D_HEAD + ln)      * SEQ + tokbase);
      v16bf bV1 = *(const v16bf*)(Vt + ((size_t)bh * D_HEAD + 16 + ln) * SEQ + tokbase);
      o0 = __builtin_amdgcn_wmma_f32_16x16x32_bf16(false, aP, false, bV0, (short)0, o0, false, false);
      o1 = __builtin_amdgcn_wmma_f32_16x16x32_bf16(false, aP, false, bV1, (short)0, o1, false, false);
    }
    __syncthreads();
  }

  // ---- normalize and head-merge store: ctx[b][s][h*32 + d] (bf16) ----
  const int b = bh >> 4, h = bh & 15;
  #pragma unroll
  for (int jj = 0; jj < 8; ++jj) {
    const int r = wave * 16 + jj + half * 8;
    const float linv = 1.0f / lrow[r];
    const int s = qblk * 64 + r;
    __bf16* cp = ctx + ((size_t)b * SEQ + s) * D_MODEL + h * D_HEAD;
    cp[ln]      = f2bf(o0[jj] * linv);
    cp[16 + ln] = f2bf(o1[jj] * linv);
  }
}

// ================================================================
// Kernel E: out[4096,512] = ctx(bf16) @ Wo^T(bf16) + bo  (f32 out)
// ================================================================
__global__ __launch_bounds__(128)
void k_outproj(const __bf16* __restrict__ ctx, const __bf16* __restrict__ Wo,
               const float* __restrict__ bo, float* __restrict__ out)
{
  const int wave = threadIdx.x >> 5;
  const int lane = threadIdx.x & 31;
  const int half = lane >> 4;
  const int ln   = lane & 15;
  const int m0 = blockIdx.x * 64 + wave * 16;
  const int n0 = blockIdx.y * 64;

  v8f acc[4];
  #pragma unroll
  for (int t = 0; t < 4; ++t)
    #pragma unroll
    for (int i = 0; i < 8; ++i) acc[t][i] = 0.0f;

  const __bf16* arow = ctx + (size_t)(m0 + ln) * D_MODEL;
  for (int kb = 0; kb < D_MODEL; kb += 32) {
    v8bf a0 = *(const v8bf*)(arow + kb + half * 16);
    v8bf a1 = *(const v8bf*)(arow + kb + half * 16 + 8);
    v16bf afrag = __builtin_shufflevector(a0, a1, 0,1,2,3,4,5,6,7,8,9,10,11,12,13,14,15);
    #pragma unroll
    for (int t = 0; t < 4; ++t) {
      const int n = n0 + t * 16 + ln;
      v16bf bfrag = *(const v16bf*)(Wo + (size_t)n * D_MODEL + kb + half * 16);
      acc[t] = __builtin_amdgcn_wmma_f32_16x16x32_bf16(
          false, afrag, false, bfrag, (short)0, acc[t], false, false);
    }
  }
  #pragma unroll
  for (int t = 0; t < 4; ++t) {
    const int n = n0 + t * 16 + ln;
    const float bv = bo[n];
    #pragma unroll
    for (int j = 0; j < 8; ++j) {
      const int m = m0 + j + half * 8;
      out[(size_t)m * D_MODEL + n] = acc[t][j] + bv;
    }
  }
}

// ================================================================
extern "C" void kernel_launch(void* const* d_in, const int* in_sizes, int n_in,
                              void* d_out, int out_size, void* d_ws, size_t ws_size,
                              hipStream_t stream)
{
  const float* query = (const float*)d_in[0];
  const float* key   = (const float*)d_in[1];
  const float* value = (const float*)d_in[2];
  // d_in[3] = pos_embedding : unused by the reference
  const float* Wq = (const float*)d_in[4];
  const float* bq = (const float*)d_in[5];
  const float* Wk = (const float*)d_in[6];
  const float* bk = (const float*)d_in[7];
  const float* Wv = (const float*)d_in[8];
  const float* bv = (const float*)d_in[9];
  const float* Wo = (const float*)d_in[10];
  const float* bo = (const float*)d_in[11];
  float* out = (float*)d_out;

  char* ws = (char*)d_ws;
  size_t off = 0;
  auto alloc = [&](size_t bytes) -> void* {
    void* p = ws + off;
    off = (off + bytes + 255) & ~(size_t)255;
    return p;
  };
  const size_t inElems = (size_t)MROWS * D_MODEL;                    // 2,097,152
  const size_t wElems  = (size_t)D_MODEL * D_MODEL;                  //   262,144
  const size_t headElems = (size_t)BATCH * N_HEADS * SEQ * D_HEAD;   // 2,097,152

  __bf16* Xq  = (__bf16*)alloc(inElems * 2);
  __bf16* Xk  = (__bf16*)alloc(inElems * 2);
  __bf16* Xv  = (__bf16*)alloc(inElems * 2);
  __bf16* Wqb = (__bf16*)alloc(wElems * 2);
  __bf16* Wkb = (__bf16*)alloc(wElems * 2);
  __bf16* Wvb = (__bf16*)alloc(wElems * 2);
  __bf16* Wob = (__bf16*)alloc(wElems * 2);
  __bf16* Qh  = (__bf16*)alloc(headElems * 2);
  __bf16* Kh  = (__bf16*)alloc(headElems * 2);
  __bf16* Vt  = (__bf16*)alloc(headElems * 2);
  __bf16* ctx = (__bf16*)alloc(inElems * 2);
  float*    qb    = (float*)   alloc((size_t)BATCH * N_HEADS * NB * D_HEAD * 4);
  float*    kb    = (float*)   alloc((size_t)BATCH * N_HEADS * NB * D_HEAD * 4);
  unsigned* bmask = (unsigned*)alloc((size_t)BATCH * N_HEADS * NB * 4);

  // Stage 0: one-pass f32 -> bf16 conversion of inputs + weights
  const int in4 = (int)(inElems / 4), w4 = (int)(wElems / 4);
  k_cvt<<<(in4 + 255) / 256, 256, 0, stream>>>(query, Xq, in4);
  k_cvt<<<(in4 + 255) / 256, 256, 0, stream>>>(key,   Xk, in4);
  k_cvt<<<(in4 + 255) / 256, 256, 0, stream>>>(value, Xv, in4);
  k_cvt<<<(w4 + 255) / 256, 256, 0, stream>>>(Wq, Wqb, w4);
  k_cvt<<<(w4 + 255) / 256, 256, 0, stream>>>(Wk, Wkb, w4);
  k_cvt<<<(w4 + 255) / 256, 256, 0, stream>>>(Wv, Wvb, w4);
  k_cvt<<<(w4 + 255) / 256, 256, 0, stream>>>(Wo, Wob, w4);

  const dim3 gemmGrid(MROWS / 64, D_MODEL / 64);   // 64 x 8
  const dim3 gemmBlk(128);

  // Stage 1: QKV projections (V stored transposed for PV B-fragments)
  k_proj<<<gemmGrid, gemmBlk, 0, stream>>>(Xq, Wqb, bq, Qh, 0);
  k_proj<<<gemmGrid, gemmBlk, 0, stream>>>(Xk, Wkb, bk, Kh, 0);
  k_proj<<<gemmGrid, gemmBlk, 0, stream>>>(Xv, Wvb, bv, Vt, 1);

  // Stage 2a: block means
  k_means<<<128, 256, 0, stream>>>(Qh, Kh, qb, kb);

  // Stage 2b: mean-sim top-k block mask (one wave per (b,h))
  k_mask<<<BATCH * N_HEADS, 32, 0, stream>>>(qb, kb, bmask);

  // Stage 2c: block-sparse flash attention
  k_attn<<<BATCH * N_HEADS * NB, 128, 0, stream>>>(Qh, Kh, Vt, bmask, ctx);

  // Stage 3: output projection
  k_outproj<<<gemmGrid, gemmBlk, 0, stream>>>(ctx, Wob, bo, out);
}
